// NeuralNetwork_83391085019611
// MI455X (gfx1250) — compile-verified, matrix operand order checked by reading, not run on hardware
//
#include <hip/hip_runtime.h>
#include <hip/hip_bf16.h>
#include <math.h>

// ---------------------------------------------------------------------------
// DEQ head: xin = x@W1^T+b1 ; y = tanh(tanh(xin)@Wz^T+bz) ;
// z* = Anderson fixed point of f(z)=tanh(z+y) ; out = tanh(z*+y)@W2^T+b2
// GEMMs use v_wmma_f32_16x16x32_f16 (f16 in, f32 accum). Anderson loop in f32.
// ---------------------------------------------------------------------------

typedef __attribute__((ext_vector_type(16))) _Float16 v16h;
typedef __attribute__((ext_vector_type(8)))  float    v8f;

#define B_ROWS   16384
#define D_IN     784
#define D_HID    200
#define D_OUTN   10
#define D_PAD    224      // 14 tiles of 16 (>=200)
#define NT       14       // N tiles
#define KC1      25       // ceil(784/32), K padded to 800 with zeros
#define KC2      7        // 224/32
#define MEMM     6        // Anderson memory
#define LAM      1e-4f
#define EPS_TOL  1e-3f

struct Scalars { float low; int done; int copy_flag; };

// ---------------------------------------------------------------------------
// Pre-swizzle weights into WMMA B-fragment order.
// ISA §7.12.2: 16-bit A 16x32 — lane L holds row m=L&15; half h (vgpr v=h/2,
// half=h&1) holds K = 2*(v&3) + 16*(v>>2) + 8*(L>>4) + (h&1).  B (32x16) uses
// the same K mapping with N = L&15.  Fragments stored lane-contiguous so the
// GEMM loads one 32B v16h per fragment.
// ---------------------------------------------------------------------------
__device__ __forceinline__ int frag_klocal(int h, int g) {
    int v = h >> 1, half = h & 1;
    return 2 * (v & 3) + 16 * (v >> 2) + 8 * g + half;
}

__global__ void make_w1frag_kernel(const float* __restrict__ W1, _Float16* __restrict__ frag) {
    int idx = blockIdx.x * blockDim.x + threadIdx.x;
    const int total = KC1 * NT * 32 * 16;
    if (idx >= total) return;
    int h    = idx & 15;
    int lane = (idx >> 4) & 31;
    int nt   = (idx >> 9) % NT;
    int kc   = idx / (512 * NT);
    int g = lane >> 4, nn = lane & 15;
    int k = kc * 32 + frag_klocal(h, g);
    int n = nt * 16 + nn;
    float val = (k < D_IN && n < D_HID) ? W1[(size_t)n * D_IN + k] : 0.0f;
    frag[idx] = (_Float16)val;
}

__global__ void make_wzfrag_kernel(const float* __restrict__ Wz, _Float16* __restrict__ frag) {
    int idx = blockIdx.x * blockDim.x + threadIdx.x;
    const int total = KC2 * NT * 32 * 16;
    if (idx >= total) return;
    int h    = idx & 15;
    int lane = (idx >> 4) & 31;
    int nt   = (idx >> 9) % NT;
    int kc   = idx / (512 * NT);
    int g = lane >> 4, nn = lane & 15;
    int k = kc * 32 + frag_klocal(h, g);
    int n = nt * 16 + nn;
    float val = (k < D_HID && n < D_HID) ? Wz[(size_t)n * D_HID + k] : 0.0f;
    frag[idx] = (_Float16)val;
}

// ---------------------------------------------------------------------------
// GEMM1: T = tanh(x @ W1^T + b1), f16 out [B, 224] (cols>=200 are zero).
// One wave per 16-row strip, all 14 N tiles held in v8f accumulators.
// ---------------------------------------------------------------------------
__global__ __launch_bounds__(256) void gemm1_kernel(
    const float* __restrict__ x, const _Float16* __restrict__ w1frag,
    const float* __restrict__ b1, _Float16* __restrict__ T)
{
    const int lane = threadIdx.x & 31;
    const int wv   = threadIdx.x >> 5;
    const int rowbase = (blockIdx.x * 8 + wv) * 16;
    const int g = lane >> 4;
    const int m = lane & 15;
    const float* xrow = x + (size_t)(rowbase + m) * D_IN;

    v8f acc[NT] = {};
#pragma unroll 1
    for (int kc = 0; kc < KC1; ++kc) {
        const int kbase = kc * 32;
        v16h a;
#pragma unroll
        for (int h = 0; h < 8; ++h) {
            int k1 = kbase + 8 * g + h;        // contiguous run 1
            int k2 = kbase + 16 + 8 * g + h;   // contiguous run 2
            a[h]     = (_Float16)(k1 < D_IN ? xrow[k1] : 0.0f);
            a[h + 8] = (_Float16)(k2 < D_IN ? xrow[k2] : 0.0f);
        }
#pragma unroll
        for (int nt = 0; nt < NT; ++nt) {
            const v16h b = *(const v16h*)(w1frag + (((size_t)kc * NT + nt) * 32 + lane) * 16);
            acc[nt] = __builtin_amdgcn_wmma_f32_16x16x32_f16(
                false, a, false, b, (short)0, acc[nt], false, false);
        }
    }
    // C layout: lane L, vgpr r -> M = r + 8*(L>>4), N = L&15
    const int nloc = lane & 15;
#pragma unroll
    for (int nt = 0; nt < NT; ++nt) {
        int ncol = nt * 16 + nloc;
        float bias = (ncol < D_HID) ? b1[ncol] : 0.0f;
#pragma unroll
        for (int r = 0; r < 8; ++r) {
            int mm = r + 8 * g;
            float v = acc[nt][r] + bias;   // padded cols: acc=0,bias=0 -> tanh(0)=0
            T[(size_t)(rowbase + mm) * D_PAD + ncol] = (_Float16)tanhf(v);
        }
    }
}

// ---------------------------------------------------------------------------
// GEMM2: y = tanh(T @ Wz^T + bz), f32 out [B, 200]
// ---------------------------------------------------------------------------
__global__ __launch_bounds__(256) void gemm2_kernel(
    const _Float16* __restrict__ T, const _Float16* __restrict__ wzfrag,
    const float* __restrict__ bz, float* __restrict__ y)
{
    const int lane = threadIdx.x & 31;
    const int wv   = threadIdx.x >> 5;
    const int rowbase = (blockIdx.x * 8 + wv) * 16;
    const int g = lane >> 4;
    const int m = lane & 15;
    const _Float16* trow = T + (size_t)(rowbase + m) * D_PAD;

    v8f acc[NT] = {};
#pragma unroll 1
    for (int kc = 0; kc < KC2; ++kc) {
        const int kbase = kc * 32;
        v16h a;
#pragma unroll
        for (int h = 0; h < 8; ++h) {
            a[h]     = trow[kbase + 8 * g + h];
            a[h + 8] = trow[kbase + 16 + 8 * g + h];
        }
#pragma unroll
        for (int nt = 0; nt < NT; ++nt) {
            const v16h b = *(const v16h*)(wzfrag + (((size_t)kc * NT + nt) * 32 + lane) * 16);
            acc[nt] = __builtin_amdgcn_wmma_f32_16x16x32_f16(
                false, a, false, b, (short)0, acc[nt], false, false);
        }
    }
    const int nloc = lane & 15;
#pragma unroll
    for (int nt = 0; nt < NT; ++nt) {
        int ncol = nt * 16 + nloc;
        if (ncol < D_HID) {
            float bias = bz[ncol];
#pragma unroll
            for (int r = 0; r < 8; ++r) {
                int mm = r + 8 * g;
                y[(size_t)(rowbase + mm) * D_HID + ncol] = tanhf(acc[nt][r] + bias);
            }
        }
    }
}

// ---------------------------------------------------------------------------
// Anderson state init: X0=0, F0=tanh(y), X1=F0, F1=tanh(F0+y), rest zero.
// ---------------------------------------------------------------------------
__global__ void init_state_kernel(const float* __restrict__ y, float* __restrict__ X,
                                  float* __restrict__ F, float* __restrict__ lx)
{
    int idx = blockIdx.x * blockDim.x + threadIdx.x;
    if (idx >= B_ROWS * D_HID) return;
    int b = idx / D_HID, j = idx % D_HID;
    float yv = y[idx];
    float f0 = tanhf(yv);
    float f1 = tanhf(f0 + yv);
    size_t base = ((size_t)b * MEMM) * D_HID + j;
    X[base + 0 * D_HID] = 0.0f; F[base + 0 * D_HID] = f0;
    X[base + 1 * D_HID] = f0;   F[base + 1 * D_HID] = f1;
#pragma unroll
    for (int i = 2; i < MEMM; ++i) { X[base + i * D_HID] = 0.0f; F[base + i * D_HID] = 0.0f; }
    lx[idx] = 0.0f;
}

__global__ void init_scalars_kernel(Scalars* sc) {
    if (threadIdx.x == 0) { sc->low = 1e8f; sc->done = 0; sc->copy_flag = 0; }
}

// ---------------------------------------------------------------------------
// One Anderson iteration: one wave32 per batch row.
// Gram of residuals -> SPD 6x6 solve (== reference bordered 7x7 solve):
//   alpha = A^{-1} mask / (mask . A^{-1} mask),  A = GG + diag(active?lam:1)
// ---------------------------------------------------------------------------
__global__ __launch_bounds__(256) void anderson_step_kernel(
    float* __restrict__ X, float* __restrict__ F, const float* __restrict__ y,
    float* __restrict__ partials, const Scalars* __restrict__ sc, int k)
{
    __shared__ float sred[16];
    if (sc->done) return;  // uniform: reduce_update also early-exits
    const int lane = threadIdx.x & 31;
    const int wv   = threadIdx.x >> 5;
    const int row  = blockIdx.x * 8 + wv;
    const int n    = (k < MEMM) ? k : MEMM;
    const int kmod = k % MEMM;

    float Fv[MEMM][7], G[MEMM][7];
#pragma unroll
    for (int t = 0; t < 7; ++t) {
        int j = lane + 32 * t;
        bool ok = j < D_HID;
#pragma unroll
        for (int i = 0; i < MEMM; ++i) {
            size_t ix = ((size_t)row * MEMM + i) * D_HID + j;
            float fv = ok ? F[ix] : 0.0f;
            float xv = ok ? X[ix] : 0.0f;
            Fv[i][t] = fv;
            G[i][t] = (i < n) ? (fv - xv) : 0.0f;
        }
    }
    float gg[21];
    {
        int p = 0;
#pragma unroll
        for (int i = 0; i < MEMM; ++i)
#pragma unroll
            for (int jj = i; jj < MEMM; ++jj) {
                float s = 0.0f;
#pragma unroll
                for (int t = 0; t < 7; ++t) s += G[i][t] * G[jj][t];
                gg[p++] = s;
            }
    }
#pragma unroll
    for (int off = 16; off >= 1; off >>= 1)
#pragma unroll
        for (int q = 0; q < 21; ++q) gg[q] += __shfl_xor(gg[q], off, 32);

    float Mm[MEMM][7];
    {
        int p = 0;
#pragma unroll
        for (int i = 0; i < MEMM; ++i)
#pragma unroll
            for (int jj = i; jj < MEMM; ++jj) { Mm[i][jj] = gg[p]; Mm[jj][i] = gg[p]; ++p; }
    }
#pragma unroll
    for (int i = 0; i < MEMM; ++i) {
        Mm[i][i] += (i < n) ? LAM : 1.0f;   // inactive rows -> identity
        Mm[i][6]  = (i < n) ? 1.0f : 0.0f;  // rhs = mask
    }
    // pivot-free Gauss-Jordan (A is SPD on active block, identity elsewhere)
#pragma unroll
    for (int c = 0; c < MEMM; ++c) {
        float inv = 1.0f / Mm[c][c];
#pragma unroll
        for (int r = 0; r < MEMM; ++r) {
            if (r == c) continue;
            float fac = Mm[r][c] * inv;
#pragma unroll
            for (int cc = 0; cc < 7; ++cc) Mm[r][cc] -= fac * Mm[c][cc];
        }
    }
    float alpha[MEMM], denom = 0.0f;
#pragma unroll
    for (int i = 0; i < MEMM; ++i) {
        float w = Mm[i][6] / Mm[i][i];
        alpha[i] = w;
        denom += ((i < n) ? 1.0f : 0.0f) * w;
    }
    float dinv = 1.0f / denom;
#pragma unroll
    for (int i = 0; i < MEMM; ++i) alpha[i] *= dinv;

    float rs = 0.0f, fs = 0.0f;
#pragma unroll
    for (int t = 0; t < 7; ++t) {
        int j = lane + 32 * t;
        if (j < D_HID) {
            float xk = 0.0f;
#pragma unroll
            for (int i = 0; i < MEMM; ++i) xk += alpha[i] * Fv[i][t];
            float fk = tanhf(xk + y[(size_t)row * D_HID + j]);
            size_t ixo = ((size_t)row * MEMM + kmod) * D_HID + j;
            X[ixo] = xk; F[ixo] = fk;
            float df = fk - xk;
            rs += df * df; fs += fk * fk;
        }
    }
#pragma unroll
    for (int off = 16; off >= 1; off >>= 1) {
        rs += __shfl_xor(rs, off, 32);
        fs += __shfl_xor(fs, off, 32);
    }
    if (lane == 0) { sred[wv] = rs; sred[8 + wv] = fs; }
    __syncthreads();
    if (threadIdx.x == 0) {
        float a = 0.0f, b = 0.0f;
#pragma unroll
        for (int w = 0; w < 8; ++w) { a += sred[w]; b += sred[8 + w]; }
        partials[2 * blockIdx.x]     = a;
        partials[2 * blockIdx.x + 1] = b;
    }
}

// Deterministic fixed-order global reduction + scalar state update.
__global__ void reduce_update_kernel(const float* __restrict__ partials, Scalars* __restrict__ sc)
{
    __shared__ float sa[256], sb[256];
    if (sc->done) { if (threadIdx.x == 0) sc->copy_flag = 0; return; }
    int tid = threadIdx.x;
    float a = 0.0f, b = 0.0f;
    for (int i = tid; i < 2048; i += 256) { a += partials[2 * i]; b += partials[2 * i + 1]; }
    sa[tid] = a; sb[tid] = b;
    __syncthreads();
    for (int s = 128; s > 0; s >>= 1) {
        if (tid < s) { sa[tid] += sa[tid + s]; sb[tid] += sb[tid + s]; }
        __syncthreads();
    }
    if (tid == 0) {
        float rel = sqrtf(sa[0]) / (1e-5f + sqrtf(sb[0]));
        int better = rel < sc->low ? 1 : 0;
        sc->copy_flag = better;
        if (better) sc->low = rel;
        if (rel < EPS_TOL) sc->done = 1;
    }
}

__global__ void copy_best_kernel(float* __restrict__ lx, const float* __restrict__ X,
                                 const Scalars* __restrict__ sc, int kmod)
{
    if (!sc->copy_flag) return;
    int idx = blockIdx.x * blockDim.x + threadIdx.x;
    if (idx >= B_ROWS * D_HID) return;
    int b = idx / D_HID, j = idx % D_HID;
    lx[idx] = X[((size_t)b * MEMM + kmod) * D_HID + j];
}

// z = tanh(lx + y) ; logits = z @ W2^T + b2  (one wave per row, N=10)
__global__ __launch_bounds__(256) void head_kernel(
    const float* __restrict__ lx, const float* __restrict__ y,
    const float* __restrict__ W2, const float* __restrict__ b2,
    float* __restrict__ out)
{
    const int lane = threadIdx.x & 31;
    const int wv   = threadIdx.x >> 5;
    const int row  = blockIdx.x * 8 + wv;
    float zv[7];
#pragma unroll
    for (int t = 0; t < 7; ++t) {
        int j = lane + 32 * t;
        zv[t] = (j < D_HID)
              ? tanhf(lx[(size_t)row * D_HID + j] + y[(size_t)row * D_HID + j])
              : 0.0f;
    }
#pragma unroll
    for (int o = 0; o < D_OUTN; ++o) {
        float s = 0.0f;
#pragma unroll
        for (int t = 0; t < 7; ++t) {
            int j = lane + 32 * t;
            if (j < D_HID) s += zv[t] * W2[o * D_HID + j];
        }
#pragma unroll
        for (int off = 16; off >= 1; off >>= 1) s += __shfl_xor(s, off, 32);
        if (lane == 0) out[(size_t)row * D_OUTN + o] = s + b2[o];
    }
}

// ---------------------------------------------------------------------------
extern "C" void kernel_launch(void* const* d_in, const int* in_sizes, int n_in,
                              void* d_out, int out_size, void* d_ws, size_t ws_size,
                              hipStream_t stream)
{
    (void)in_sizes; (void)n_in; (void)out_size; (void)ws_size;
    const float* x  = (const float*)d_in[0];
    const float* W1 = (const float*)d_in[1];
    const float* b1 = (const float*)d_in[2];
    const float* Wz = (const float*)d_in[3];
    const float* bz = (const float*)d_in[4];
    const float* W2 = (const float*)d_in[5];
    const float* b2 = (const float*)d_in[6];
    float* out = (float*)d_out;

    char* ws = (char*)d_ws;
    size_t off = 0;
    auto take = [&](size_t bytes) -> void* {
        void* p = ws + off;
        off += (bytes + 255) & ~(size_t)255;
        return p;
    };
    _Float16* w1frag = (_Float16*)take((size_t)KC1 * NT * 32 * 16 * 2);   // 350 KB
    _Float16* wzfrag = (_Float16*)take((size_t)KC2 * NT * 32 * 16 * 2);   //  98 KB
    _Float16* T      = (_Float16*)take((size_t)B_ROWS * D_PAD * 2);       // 7.0 MB
    float* ybuf      = (float*)take((size_t)B_ROWS * D_HID * 4);          // 12.5 MB
    float* Xbuf      = (float*)take((size_t)B_ROWS * MEMM * D_HID * 4);   // 75 MB
    float* Fbuf      = (float*)take((size_t)B_ROWS * MEMM * D_HID * 4);   // 75 MB
    float* lx        = (float*)take((size_t)B_ROWS * D_HID * 4);          // 12.5 MB
    float* partials  = (float*)take((size_t)2048 * 2 * 4);
    Scalars* sc      = (Scalars*)take(256);

    make_w1frag_kernel<<<(KC1 * NT * 512 + 255) / 256, 256, 0, stream>>>(W1, w1frag);
    make_wzfrag_kernel<<<(KC2 * NT * 512 + 255) / 256, 256, 0, stream>>>(Wz, wzfrag);
    gemm1_kernel<<<B_ROWS / 128, 256, 0, stream>>>(x, w1frag, b1, T);
    gemm2_kernel<<<B_ROWS / 128, 256, 0, stream>>>(T, wzfrag, bz, ybuf);
    init_scalars_kernel<<<1, 32, 0, stream>>>(sc);
    init_state_kernel<<<(B_ROWS * D_HID + 255) / 256, 256, 0, stream>>>(ybuf, Xbuf, Fbuf, lx);

    for (int k = 2; k < 50; ++k) {  // fixed 48 launches; device 'done' flag no-ops the tail
        anderson_step_kernel<<<B_ROWS / 8, 256, 0, stream>>>(Xbuf, Fbuf, ybuf, partials, sc, k);
        reduce_update_kernel<<<1, 256, 0, stream>>>(partials, sc);
        copy_best_kernel<<<(B_ROWS * D_HID + 255) / 256, 256, 0, stream>>>(lx, Xbuf, sc, k % MEMM);
    }
    head_kernel<<<B_ROWS / 8, 256, 0, stream>>>(lx, ybuf, W2, b2, out);
}
